// SimilarityModel_44796508897555
// MI455X (gfx1250) — matrix operand order repaired and checked by reading, not compiled, not run on hardware
//
#include <hip/hip_runtime.h>
#include <hip/hip_bf16.h>
#include <math.h>

// Problem constants (fixed by the reference): B=64, T=4096, D=1024.
#define BATCH 64
#define TSEQ  4096
#define DDIM  1024

typedef float    v4f  __attribute__((ext_vector_type(4)));
typedef float    v8f  __attribute__((ext_vector_type(8)));
typedef _Float16 v16h __attribute__((ext_vector_type(16)));
typedef __fp16   h2   __attribute__((ext_vector_type(2)));   // cvt_pkrtz result type

union APack { v16h v; h2 p[8]; };

// ---------------------------------------------------------------------------
// Pass 1: sim_raw[b][t] = dot(video[b][t], lang_f16[b]) / max(||video[b][t]||, 1e-12)
// Lang normalization is applied in pass 2 (positive per-batch scalar).
// Block = 256 threads (8 waves). Wave w covers 16 rows. Grid = B * (T/128).
// ---------------------------------------------------------------------------
__global__ __launch_bounds__(256) void sim_pass1(const float* __restrict__ video,
                                                 const float* __restrict__ lang,
                                                 float* __restrict__ sim_out)
{
    __shared__ __align__(64) _Float16 lds_lang[DDIM]; // 2 KB, f16 lang vector

    const int b       = blockIdx.x >> 5;      // 32 blocks per batch
    const int tileBlk = blockIdx.x & 31;      // 128 rows per block
    const int tid     = threadIdx.x;
    const int wave    = tid >> 5;
    const int lane    = tid & 31;
    const int row     = lane & 15;
    const int hi      = lane >> 4;            // 0: K 0-7/16-23, 1: K 8-15/24-31

    // Stage lang (f32 -> f16) into LDS: 256 threads x 4 floats.
    {
        const v4f lv = *(const v4f*)(lang + (size_t)b * DDIM + tid * 4);
        *(h2*)&lds_lang[tid * 4 + 0] = __builtin_amdgcn_cvt_pkrtz(lv.x, lv.y);
        *(h2*)&lds_lang[tid * 4 + 2] = __builtin_amdgcn_cvt_pkrtz(lv.z, lv.w);
    }
    __syncthreads();

    const int rowBase = tileBlk * 128 + wave * 16;
    const float* vptr = video + ((size_t)b * TSEQ + rowBase + row) * DDIM;

    v8f   c   = {};
    float ssq = 0.0f;

#pragma unroll 4
    for (int k0 = 0; k0 < DDIM; k0 += 32) {
        // A-fragment loads (documented 16-bit A 16x32 layout), f32 source:
        // lane(hi=0): K k0+0..7 and k0+16..23 ; lane(hi=1): +8 on both.
        const float* p = vptr + k0 + hi * 8;
        v4f a0 = *(const v4f*)(p + 0);
        v4f a1 = *(const v4f*)(p + 4);
        v4f a2 = *(const v4f*)(p + 16);
        v4f a3 = *(const v4f*)(p + 20);

        // Per-lane sum of squares (lane pair covers the full row).
        ssq = fmaf(a0.x, a0.x, ssq); ssq = fmaf(a0.y, a0.y, ssq);
        ssq = fmaf(a0.z, a0.z, ssq); ssq = fmaf(a0.w, a0.w, ssq);
        ssq = fmaf(a1.x, a1.x, ssq); ssq = fmaf(a1.y, a1.y, ssq);
        ssq = fmaf(a1.z, a1.z, ssq); ssq = fmaf(a1.w, a1.w, ssq);
        ssq = fmaf(a2.x, a2.x, ssq); ssq = fmaf(a2.y, a2.y, ssq);
        ssq = fmaf(a2.z, a2.z, ssq); ssq = fmaf(a2.w, a2.w, ssq);
        ssq = fmaf(a3.x, a3.x, ssq); ssq = fmaf(a3.y, a3.y, ssq);
        ssq = fmaf(a3.z, a3.z, ssq); ssq = fmaf(a3.w, a3.w, ssq);

        // Pack f32 -> f16 in A-fragment element order.
        APack ap;
        ap.p[0] = __builtin_amdgcn_cvt_pkrtz(a0.x, a0.y);
        ap.p[1] = __builtin_amdgcn_cvt_pkrtz(a0.z, a0.w);
        ap.p[2] = __builtin_amdgcn_cvt_pkrtz(a1.x, a1.y);
        ap.p[3] = __builtin_amdgcn_cvt_pkrtz(a1.z, a1.w);
        ap.p[4] = __builtin_amdgcn_cvt_pkrtz(a2.x, a2.y);
        ap.p[5] = __builtin_amdgcn_cvt_pkrtz(a2.z, a2.w);
        ap.p[6] = __builtin_amdgcn_cvt_pkrtz(a3.x, a3.y);
        ap.p[7] = __builtin_amdgcn_cvt_pkrtz(a3.z, a3.w);

        // B-fragment: lang chunk replicated across all 16 columns, so the
        // result is independent of the N->lane mapping. K split matches A:
        // lane(hi=0): K k0+0..7 and k0+16..23 ; lane(hi=1): +8 on both.
        APack bp;
#pragma unroll
        for (int q = 0; q < 4; ++q) {
            bp.p[q]     = *(const h2*)&lds_lang[k0 + hi * 8 + 2 * q];
            bp.p[q + 4] = *(const h2*)&lds_lang[k0 + 16 + hi * 8 + 2 * q];
        }
        v16h bv = bp.v;

        c = __builtin_amdgcn_wmma_f32_16x16x32_f16(
                false, ap.v, false, bv, (short)0, c, false, false);
    }

    // Combine the lane-pair partial sums of squares: row r lives in lanes r, r+16.
    float ssqTot = ssq + __shfl_xor(ssq, 16, 32);

    // D layout: lanes 0-15 hold rows 0-7 in c[0..7]; lanes 16-31 hold rows 8-15.
    float r[8];
#pragma unroll
    for (int j = 0; j < 8; ++j) {
        float st    = __shfl(ssqTot, (hi << 3) + j, 32);  // ssq of row hi*8+j
        float denom = fmaxf(sqrtf(st), 1e-12f);
        r[j]        = c[j] / denom;
    }

    if ((lane & 15) == 0) {
        float* dst = sim_out + (size_t)b * TSEQ + rowBase + (hi << 3);
        v4f o0 = { r[0], r[1], r[2], r[3] };
        v4f o1 = { r[4], r[5], r[6], r[7] };
        *(v4f*)dst       = o0;
        *((v4f*)dst + 1) = o1;
    }
}

// ---------------------------------------------------------------------------
// Pass 2: per-batch lang norm, argmax (first occurrence), boundary logic.
// One block per batch; sims staged in LDS (16 KB, fed from L2-resident ws).
// ---------------------------------------------------------------------------
__global__ __launch_bounds__(256) void bounds_pass2(const float* __restrict__ lang,
                                                    const float* __restrict__ sims,
                                                    const float* __restrict__ delta_p,
                                                    float* __restrict__ out)
{
    __shared__ float ssim[TSEQ];
    __shared__ float rv[256];
    __shared__ int   ri[256];
    __shared__ float s_inv;

    const int b   = blockIdx.x;
    const int tid = threadIdx.x;
    const float delta = *delta_p;

    // Lang norm.
    float ss = 0.0f;
    for (int i = tid; i < DDIM; i += 256) {
        float v = lang[(size_t)b * DDIM + i];
        ss = fmaf(v, v, ss);
    }
    rv[tid] = ss;
    __syncthreads();
    for (int s = 128; s > 0; s >>= 1) {
        if (tid < s) rv[tid] += rv[tid + s];
        __syncthreads();
    }
    if (tid == 0) s_inv = 1.0f / fmaxf(sqrtf(rv[0]), 1e-12f);
    __syncthreads();
    const float inv_l = s_inv;

    // Scale sims, stage in LDS, per-thread first-occurrence argmax.
    float bv = -INFINITY;
    int   bi = TSEQ;
    for (int t = tid; t < TSEQ; t += 256) {
        float v = sims[(size_t)b * TSEQ + t] * inv_l;
        ssim[t] = v;
        if (v > bv) { bv = v; bi = t; }   // keeps earliest index (ascending t)
    }
    rv[tid] = bv; ri[tid] = bi;
    __syncthreads();
    for (int s = 128; s > 0; s >>= 1) {
        if (tid < s) {
            float ov = rv[tid + s]; int oi = ri[tid + s];
            if (ov > rv[tid] || (ov == rv[tid] && oi < ri[tid])) {
                rv[tid] = ov; ri[tid] = oi;
            }
        }
        __syncthreads();
    }
    const int   peak = ri[0];
    const float pv   = rv[0];
    __syncthreads();

    // last bad index before peak / first bad index after peak.
    int lastb = -1, firstb = TSEQ;
    for (int t = tid; t < TSEQ; t += 256) {
        bool bad = fabsf(ssim[t] - pv) > delta;
        if (bad) {
            if (t < peak && t > lastb)  lastb  = t;
            if (t > peak && t < firstb) firstb = t;
        }
    }
    ri[tid] = lastb;
    __syncthreads();
    for (int s = 128; s > 0; s >>= 1) {
        if (tid < s) ri[tid] = max(ri[tid], ri[tid + s]);
        __syncthreads();
    }
    const int last_bad = ri[0];
    __syncthreads();
    ri[tid] = firstb;
    __syncthreads();
    for (int s = 128; s > 0; s >>= 1) {
        if (tid < s) ri[tid] = min(ri[tid], ri[tid + s]);
        __syncthreads();
    }
    const int first_bad = ri[0];

    if (tid == 0) {
        int s0  = last_bad + 1;
        int st  = (s0 == peak) ? s0 : s0 + 1;
        int e0  = first_bad - 1;
        int en  = (e0 == peak) ? e0 : e0 - 1;
        out[b]         = (float)st;
        out[BATCH + b] = (float)en;
    }
}

extern "C" void kernel_launch(void* const* d_in, const int* in_sizes, int n_in,
                              void* d_out, int out_size, void* d_ws, size_t ws_size,
                              hipStream_t stream) {
    const float* video = (const float*)d_in[0];   // (64, 4096, 1024) f32
    const float* lang  = (const float*)d_in[1];   // (64, 1, 1024) f32
    const float* delta = (const float*)d_in[2];   // scalar f32
    float* out  = (float*)d_out;                  // 128 floats: start[64] ++ end[64]
    float* sims = (float*)d_ws;                   // 64*4096 f32 = 1 MB scratch

    sim_pass1<<<BATCH * (TSEQ / 128), 256, 0, stream>>>(video, lang, sims);
    bounds_pass2<<<BATCH, 256, 0, stream>>>(lang, sims, delta, out);
}
